// H2GCNNet_62423054680289
// MI455X (gfx1250) — matrix operand order repaired and checked by reading, not compiled, not run on hardware
//
#include <hip/hip_runtime.h>
#include <hip/hip_bf16.h>

typedef __attribute__((ext_vector_type(16))) __bf16 v16bf;
typedef __attribute__((ext_vector_type(8)))  float  v8f;

#define IN_C 500
#define HID  128
#define OUTC 40
#define KCH1 16   /* ceil(500/32) K-chunks for GEMM1 */
#define KCH2 12   /* 384/32 K-chunks for GEMM2 */
#define NT1  8    /* 128/16 column tiles */
#define NT2  3    /* 48/16 column tiles (40 padded to 48) */

// ---------------------------------------------------------------------------
// Pre-pack W1 [500x128] f32 -> bf16 in WMMA B-register layout.
// B (32x16 bf16) layout per wave32: lane l holds 16 bf16; half=l>>4,
// element e -> K = half*8 + e (e<8), 8 + half*8 + e (e>=8), column = (l&15). Stored so a lane reads its
// 16 elements contiguously: offset = ((ch*NT1 + nt)*32 + lane)*16 + e.
// ---------------------------------------------------------------------------
__global__ void pack_W1_kernel(const float* __restrict__ W1, __bf16* __restrict__ W1p) {
    int idx = blockIdx.x * blockDim.x + threadIdx.x;
    if (idx >= KCH1 * NT1 * 32 * 16) return;
    int e    = idx & 15;
    int lane = (idx >> 4) & 31;
    int nt   = (idx >> 9) & 7;
    int ch   = idx >> 12;
    int half = lane >> 4;
    int k    = ch * 32 + ((e < 8) ? (half * 8 + e) : (16 + half * 8 + (e - 8)));
    int col  = nt * 16 + (lane & 15);
    float v  = (k < IN_C) ? W1[k * HID + col] : 0.0f;
    W1p[idx] = (__bf16)v;
}

// Same packing for Wf [384x40] f32 -> bf16, columns padded to 48 with zeros.
__global__ void pack_Wf_kernel(const float* __restrict__ Wf, __bf16* __restrict__ Wfp) {
    int idx = blockIdx.x * blockDim.x + threadIdx.x;
    if (idx >= KCH2 * NT2 * 32 * 16) return;
    int e    = idx & 15;
    int lane = (idx >> 4) & 31;
    int rest = idx >> 9;
    int nt   = rest % 3;
    int ch   = rest / 3;
    int half = lane >> 4;
    int k    = ch * 32 + ((e < 8) ? (half * 8 + e) : (16 + half * 8 + (e - 8)));
    int col  = nt * 16 + (lane & 15);
    float v  = (col < OUTC) ? Wf[k * OUTC + col] : 0.0f;
    Wfp[idx] = (__bf16)v;
}

// ---------------------------------------------------------------------------
// GEMM1 + ReLU: h = relu(x @ W1).  One block (8 waves) per 16-row tile;
// wave w computes the 16x16 output tile for columns [16w, 16w+16).
// A (16x32 bf16) per-lane layout: m = lane&15 (row), half = lane>>4,
// e<8 -> K = k0 + half*8 + e ; e>=8 -> K = k0 + 16 + half*8 + (e-8).
// Also zero-initializes h1/h2 (each element written exactly once).
// ---------------------------------------------------------------------------
__global__ void gemm1_relu_kernel(const float* __restrict__ x,
                                  const __bf16* __restrict__ W1p,
                                  float* __restrict__ h,
                                  float* __restrict__ h1,
                                  float* __restrict__ h2, int N) {
    int lane = threadIdx.x & 31;
    int wave = threadIdx.x >> 5;
    int half = lane >> 4;
    int mrow = blockIdx.x * 16 + (lane & 15);
    int arow = (mrow < N) ? mrow : (N - 1);           // clamp loads, guard stores
    const float* xr = x + (size_t)arow * IN_C;

    v8f acc = {0.f, 0.f, 0.f, 0.f, 0.f, 0.f, 0.f, 0.f};

    for (int ch = 0; ch < KCH1; ++ch) {
        int k0 = ch * 32;
        v16bf a;
        // region e = 0..7
        int ks = k0 + half * 8;
        if (ks + 8 <= IN_C) {
            float4 f0 = *(const float4*)(xr + ks);
            float4 f1 = *(const float4*)(xr + ks + 4);
            a[0] = (__bf16)f0.x; a[1] = (__bf16)f0.y; a[2] = (__bf16)f0.z; a[3] = (__bf16)f0.w;
            a[4] = (__bf16)f1.x; a[5] = (__bf16)f1.y; a[6] = (__bf16)f1.z; a[7] = (__bf16)f1.w;
        } else {
            for (int e = 0; e < 8; ++e) {
                int k = ks + e;
                a[e] = (__bf16)((k < IN_C) ? xr[k] : 0.0f);
            }
        }
        // region e = 8..15
        int ks2 = k0 + 16 + half * 8;
        if (ks2 + 8 <= IN_C) {
            float4 f2 = *(const float4*)(xr + ks2);
            float4 f3 = *(const float4*)(xr + ks2 + 4);
            a[8]  = (__bf16)f2.x; a[9]  = (__bf16)f2.y; a[10] = (__bf16)f2.z; a[11] = (__bf16)f2.w;
            a[12] = (__bf16)f3.x; a[13] = (__bf16)f3.y; a[14] = (__bf16)f3.z; a[15] = (__bf16)f3.w;
        } else {
            for (int e = 0; e < 8; ++e) {
                int k = ks2 + e;
                a[8 + e] = (__bf16)((k < IN_C) ? xr[k] : 0.0f);
            }
        }
        // packed B tile: two b128 loads
        v16bf b = *(const v16bf*)(W1p + ((size_t)(ch * NT1 + wave) * 32 + lane) * 16);

        acc = __builtin_amdgcn_wmma_f32_16x16x32_bf16(
            false, a, false, b, (short)0, acc, false, false);
    }

    // D layout: lane column n = lane&15, VGPR j -> row j + 8*half
    int col = wave * 16 + (lane & 15);
    #pragma unroll
    for (int j = 0; j < 8; ++j) {
        int r = blockIdx.x * 16 + j + 8 * half;
        if (r < N) {
            float v = acc[j];
            v = v > 0.0f ? v : 0.0f;
            size_t o = (size_t)r * HID + col;
            h[o]  = v;
            h1[o] = 0.0f;
            h2[o] = 0.0f;
        }
    }
}

// ---------------------------------------------------------------------------
// Native non-returning f32 atomic adds (STOREcnt-tracked, device scope so the
// RMW executes in L2's atomic units, where h1/h2 are resident). Inline asm
// guarantees global_atomic_add_f32 instead of a CAS-loop expansion.
// Four consecutive columns share one 64-bit address via the IOFFSET field.
// ---------------------------------------------------------------------------
__device__ __forceinline__ void atomic_add_f32x4(float* p, float a, float b,
                                                 float c, float d) {
    asm volatile(
        "global_atomic_add_f32 %0, %1, off scope:SCOPE_DEV\n\t"
        "global_atomic_add_f32 %0, %2, off offset:4 scope:SCOPE_DEV\n\t"
        "global_atomic_add_f32 %0, %3, off offset:8 scope:SCOPE_DEV\n\t"
        "global_atomic_add_f32 %0, %4, off offset:12 scope:SCOPE_DEV"
        :
        : "v"(p), "v"(a), "v"(b), "v"(c), "v"(d)
        : "memory");
}

// ---------------------------------------------------------------------------
// SpMM: out[row] += val * h[col], edge-parallel, one wave per edge.
// 32 lanes x float4 cover the 128-wide row: coalesced 512B gather (L2-resident
// h) + coalesced f32 atomics (L2-resident accumulator). Prefetch next gather.
// ---------------------------------------------------------------------------
__global__ void spmm_atomic_kernel(const float* __restrict__ val,
                                   const int* __restrict__ row,
                                   const int* __restrict__ col,
                                   const float* __restrict__ h,
                                   float* __restrict__ out, int E, int nwaves) {
    int lane = threadIdx.x & 31;
    int gw   = (blockIdx.x * blockDim.x + threadIdx.x) >> 5;

    for (int e = gw; e < E; e += nwaves) {
        int   r = row[e];
        int   c = col[e];
        float v = val[e];
        if (e + nwaves < E) {
            int cn = col[e + nwaves];
            __builtin_prefetch(h + (size_t)cn * HID + lane * 4, 0, 1);
        }
        float4 hv = *(const float4*)(h + (size_t)c * HID + lane * 4);
        float* o  = out + (size_t)r * HID + lane * 4;
        atomic_add_f32x4(o, v * hv.x, v * hv.y, v * hv.z, v * hv.w);
    }
}

// ---------------------------------------------------------------------------
// GEMM2 + log_softmax: out = log_softmax([h|h1|h2] @ Wf).
// One wave (blockDim=32) per 16-row tile. K=384 = 12 chunks; each chunk lies
// entirely in h, h1 or h2. Three 16x16 WMMA column tiles cover 48 (>=40) cols.
// D tiles staged in LDS, per-row log-sum-exp, coalesced contiguous store.
// ---------------------------------------------------------------------------
__global__ void gemm2_logsoftmax_kernel(const float* __restrict__ h,
                                        const float* __restrict__ h1,
                                        const float* __restrict__ h2,
                                        const __bf16* __restrict__ Wfp,
                                        float* __restrict__ out, int N) {
    __shared__ float lds[16][52];

    int lane = threadIdx.x;          // 0..31 (one wave)
    int half = lane >> 4;
    int rowbase = blockIdx.x * 16;
    int mrow = rowbase + (lane & 15);
    int arow = (mrow < N) ? mrow : (N - 1);

    v8f acc0 = {0.f,0.f,0.f,0.f,0.f,0.f,0.f,0.f};
    v8f acc1 = acc0;
    v8f acc2 = acc0;

    #pragma unroll
    for (int ch = 0; ch < KCH2; ++ch) {
        const float* src = (ch < 4) ? h : ((ch < 8) ? h1 : h2);
        int koff = (ch * 32) & 127;
        const float* ar = src + (size_t)arow * HID + koff;

        v16bf a;
        int ks  = half * 8;
        float4 f0 = *(const float4*)(ar + ks);
        float4 f1 = *(const float4*)(ar + ks + 4);
        a[0] = (__bf16)f0.x; a[1] = (__bf16)f0.y; a[2] = (__bf16)f0.z; a[3] = (__bf16)f0.w;
        a[4] = (__bf16)f1.x; a[5] = (__bf16)f1.y; a[6] = (__bf16)f1.z; a[7] = (__bf16)f1.w;
        int ks2 = 16 + half * 8;
        float4 f2 = *(const float4*)(ar + ks2);
        float4 f3 = *(const float4*)(ar + ks2 + 4);
        a[8]  = (__bf16)f2.x; a[9]  = (__bf16)f2.y; a[10] = (__bf16)f2.z; a[11] = (__bf16)f2.w;
        a[12] = (__bf16)f3.x; a[13] = (__bf16)f3.y; a[14] = (__bf16)f3.z; a[15] = (__bf16)f3.w;

        v16bf b0 = *(const v16bf*)(Wfp + ((size_t)(ch * NT2 + 0) * 32 + lane) * 16);
        v16bf b1 = *(const v16bf*)(Wfp + ((size_t)(ch * NT2 + 1) * 32 + lane) * 16);
        v16bf b2 = *(const v16bf*)(Wfp + ((size_t)(ch * NT2 + 2) * 32 + lane) * 16);

        acc0 = __builtin_amdgcn_wmma_f32_16x16x32_bf16(false, a, false, b0, (short)0, acc0, false, false);
        acc1 = __builtin_amdgcn_wmma_f32_16x16x32_bf16(false, a, false, b1, (short)0, acc1, false, false);
        acc2 = __builtin_amdgcn_wmma_f32_16x16x32_bf16(false, a, false, b2, (short)0, acc2, false, false);
    }

    // spill D tiles to LDS: lane column n = lane&15, VGPR j -> row j + 8*half
    int n = lane & 15;
    #pragma unroll
    for (int j = 0; j < 8; ++j) {
        int r = j + 8 * half;
        lds[r][n]      = acc0[j];
        lds[r][16 + n] = acc1[j];
        lds[r][32 + n] = acc2[j];
    }
    __syncthreads();

    if (lane < 16) {
        float mx = -1e30f;
        for (int c = 0; c < OUTC; ++c) mx = fmaxf(mx, lds[lane][c]);
        float s = 0.0f;
        for (int c = 0; c < OUTC; ++c) s += __expf(lds[lane][c] - mx);
        lds[lane][48] = mx + __logf(s);     // log-sum-exp bias for this row
    }
    __syncthreads();

    for (int idx = lane; idx < 16 * OUTC; idx += 32) {
        int r = idx / OUTC;
        int c = idx - r * OUTC;
        if (rowbase + r < N)
            out[(size_t)rowbase * OUTC + idx] = lds[r][c] - lds[r][48];
    }
}

// ---------------------------------------------------------------------------
extern "C" void kernel_launch(void* const* d_in, const int* in_sizes, int n_in,
                              void* d_out, int out_size, void* d_ws, size_t ws_size,
                              hipStream_t stream) {
    (void)n_in; (void)out_size; (void)ws_size;

    const float* x    = (const float*)d_in[0];
    const float* W1   = (const float*)d_in[1];
    const float* Wf   = (const float*)d_in[2];
    const float* val1 = (const float*)d_in[3];
    const float* val2 = (const float*)d_in[4];
    const int*   row1 = (const int*)d_in[5];
    const int*   col1 = (const int*)d_in[6];
    const int*   row2 = (const int*)d_in[7];
    const int*   col2 = (const int*)d_in[8];
    float* out = (float*)d_out;

    int N = in_sizes[0] / IN_C;
    int E = in_sizes[3];

    char*  ws     = (char*)d_ws;
    size_t hbytes = (size_t)N * HID * sizeof(float);
    float*  h   = (float*)ws;
    float*  h1  = (float*)(ws + hbytes);
    float*  h2  = (float*)(ws + 2 * hbytes);
    __bf16* W1p = (__bf16*)(ws + 3 * hbytes);                 // 32B-aligned (hbytes % 32 == 0)
    __bf16* Wfp = W1p + (size_t)KCH1 * NT1 * 32 * 16;

    int nW1 = KCH1 * NT1 * 32 * 16;
    int nWf = KCH2 * NT2 * 32 * 16;
    pack_W1_kernel<<<(nW1 + 255) / 256, 256, 0, stream>>>(W1, W1p);
    pack_Wf_kernel<<<(nWf + 255) / 256, 256, 0, stream>>>(Wf, Wfp);

    int mtiles = (N + 15) / 16;
    gemm1_relu_kernel<<<mtiles, 256, 0, stream>>>(x, W1p, h, h1, h2, N);

    int blocks = 2048;
    int nwaves = blocks * 256 / 32;
    spmm_atomic_kernel<<<blocks, 256, 0, stream>>>(val1, row1, col1, h, h1, E, nwaves);
    spmm_atomic_kernel<<<blocks, 256, 0, stream>>>(val2, row2, col2, h, h2, E, nwaves);

    gemm2_logsoftmax_kernel<<<mtiles, 32, 0, stream>>>(h, h1, h2, Wfp, out, N);
}
